// MultiHeadAttention_10204842295768
// MI455X (gfx1250) — matrix operand order repaired and checked by reading, not compiled, not run on hardware
//
#include <hip/hip_runtime.h>

typedef _Float16 h16;
typedef __attribute__((ext_vector_type(16))) _Float16 v16h;
typedef __attribute__((ext_vector_type(8)))  _Float16 v8h;
typedef __attribute__((ext_vector_type(4)))  _Float16 v4h;
typedef __attribute__((ext_vector_type(8)))  float    v8f;

#define TT 2048
#define DD 1024
#define HH 16
#define DK 64

union U16h { v16h v; v8h h[2]; };

// A-fragment (16-bit 16x32, ISA 7.12.2): lane l -> row r=l&15, group g=l>>4.
// halves 0..7 = K[kbase + g*8 .. +7], halves 8..15 = K[kbase + 16 + g*8 .. +7]
static __device__ __forceinline__ v16h frag_a(const h16* p, int row, int stride, int kbase, int g) {
  U16h u;
  u.h[0] = *(const v8h*)(p + (size_t)row * stride + kbase + g * 8);
  u.h[1] = *(const v8h*)(p + (size_t)row * stride + kbase + g * 8 + 16);
  return u.v;
}
// B-fragment (16-bit 32x16): lane l -> N r=l&15, group g=l>>4; halves 0..7 = K[kbase + g*8 .. +7], halves 8..15 = K[kbase + 16 + g*8 .. +7]
static __device__ __forceinline__ v16h frag_b(const h16* p, int row, int stride, int kbase, int g) {
  U16h u;
  u.h[0] = *(const v8h*)(p + (size_t)row * stride + kbase + g * 8);
  u.h[1] = *(const v8h*)(p + (size_t)row * stride + kbase + g * 8 + 16);
  return u.v;
}
static __device__ __forceinline__ v8f wmma32(v16h a, v16h b, v8f c) {
  return __builtin_amdgcn_wmma_f32_16x16x32_f16(false, a, false, b, (short)0, c, false, false);
}

// gfx1250 async global->LDS copy (ASYNCcnt-tracked), per CDNA5 ISA.
// LDS operand = addr[31:0] of the generic shared pointer (aperture rule: low 32 bits = LDS offset).
static __device__ __forceinline__ void async_ld16(h16* lds_dst, const h16* gsrc) {
  unsigned l = (unsigned)(uintptr_t)lds_dst;
  unsigned long long a = (unsigned long long)(uintptr_t)gsrc;
  asm volatile("global_load_async_to_lds_b128 %0, %1, off" :: "v"(l), "v"(a) : "memory");
}
static __device__ __forceinline__ void wait_async0() {
  asm volatile("s_wait_asynccnt 0x0" ::: "memory");
}

// ---------------- QKV projection: dst = x @ W^T (templated, branch-free) ----------------
// WHICH: 0 = Q (scaled 1/sqrt(dk), [bh][t][dk]); 1 = K ([bh][t][dk]); 2 = V^T ([bh][dk][t])
template <int WHICH>
__global__ __launch_bounds__(128) void qkv_kernel(
    const float* __restrict__ x, const float* __restrict__ W, h16* __restrict__ dst)
{
  __shared__ __align__(16) h16 xs[2][64 * 32];
  __shared__ __align__(16) h16 wsh[2][64 * 32];
  const int tid = threadIdx.x;
  const int lane = tid & 31, wave = tid >> 5;
  const int g = lane >> 4, r = lane & 15;
  const int m0 = blockIdx.x * 64;
  const int n0 = blockIdx.y * 64;

  float4 xr[4], wr[4];
  auto load_regs = [&](int kb) {
#pragma unroll
    for (int i = 0; i < 4; ++i) {
      int idx = tid + i * 128, row = idx >> 3, c4 = idx & 7;
      xr[i] = *(const float4*)(x + (size_t)(m0 + row) * DD + kb + c4 * 4);
      wr[i] = *(const float4*)(W + (size_t)(n0 + row) * DD + kb + c4 * 4);
    }
  };
  auto store_lds = [&](int buf) {
#pragma unroll
    for (int i = 0; i < 4; ++i) {
      int idx = tid + i * 128, row = idx >> 3, c4 = idx & 7;
      v4h xh = { (h16)xr[i].x, (h16)xr[i].y, (h16)xr[i].z, (h16)xr[i].w };
      *(v4h*)(&xs[buf][row * 32 + c4 * 4]) = xh;
      v4h wh = { (h16)wr[i].x, (h16)wr[i].y, (h16)wr[i].z, (h16)wr[i].w };
      *(v4h*)(&wsh[buf][row * 32 + c4 * 4]) = wh;
    }
  };

  v8f acc[4] = {};
  load_regs(0);
  store_lds(0);
  __syncthreads();
  int cur = 0;
  for (int s = 0; s < 32; ++s) {        // K loop: 32 steps of 32
    if (s < 31) load_regs((s + 1) * 32);               // overlap with WMMAs below
    v16h a = frag_a(&xs[cur][0], wave * 16 + r, 32, 0, g);
#pragma unroll
    for (int nt = 0; nt < 4; ++nt)
      acc[nt] = wmma32(a, frag_b(&wsh[cur][0], nt * 16 + r, 32, 0, g), acc[nt]);
    if (s < 31) store_lds(cur ^ 1);
    __syncthreads();
    cur ^= 1;
  }

  const int b  = m0 >> 11;                        // batch (tile never crosses)
  const int t0 = (m0 & (TT - 1)) + wave * 16 + g * 8;
  const int h  = n0 >> 6;                         // head (tile never crosses)
  if (WHICH == 2) {
    size_t base = ((size_t)(b * HH + h) * DK + r) * TT + t0;
#pragma unroll
    for (int nt = 0; nt < 4; ++nt)
#pragma unroll
      for (int v = 0; v < 8; ++v)
        dst[base + (size_t)nt * 16 * TT + v] = (h16)acc[nt][v];
  } else {
    const float sc = (WHICH == 0) ? 0.125f : 1.0f;  // fold 1/sqrt(dk) into Q
    size_t base = ((size_t)(b * HH + h) * TT + t0) * DK + r;
#pragma unroll
    for (int v = 0; v < 8; ++v)
#pragma unroll
      for (int nt = 0; nt < 4; ++nt)
        dst[base + (size_t)v * DK + nt * 16] = (h16)(acc[nt][v] * sc);
  }
}

// ---------------- Flash attention: 64-query tile per block, one (b,h) --------------
// Double-buffered K/V tiles with async global->LDS copies overlapping compute.
__global__ __launch_bounds__(128) void attn_kernel(
    const h16* __restrict__ Q, const h16* __restrict__ Kc,
    const h16* __restrict__ Vt, h16* __restrict__ Cx)
{
  __shared__ __align__(16) h16 Ks[2][64 * 64];    // [buf][key][dk]
  __shared__ __align__(16) h16 Vs[2][64 * 64];    // [buf][dk][key]
  __shared__ __align__(16) h16 Ps[4 * 16 * 64];   // per-wave P staging [q][key]
  const int tid = threadIdx.x;
  const int lane = tid & 31, wave = tid >> 5;
  const int g = lane >> 4, r = lane & 15;
  const int qblk = blockIdx.x;                    // 0..31
  const int bh   = blockIdx.y;                    // 0..63
  const int b = bh >> 4, h = bh & 15;
  const size_t kbase = (size_t)bh * TT * DK;
  const size_t vbase = (size_t)bh * DK * TT;
  const int q0 = qblk * 64 + wave * 16;

  auto issue_tile = [&](int jb, int buf) {
#pragma unroll
    for (int i = 0; i < 4; ++i) {
      int idx = tid + i * 128;                    // 0..511 -> (row, 8-half chunk)
      int row = idx >> 3, c8 = idx & 7;
      async_ld16(&Ks[buf][row * 64 + c8 * 8],
                 Kc + kbase + (size_t)(jb * 64 + row) * DK + c8 * 8);
      async_ld16(&Vs[buf][row * 64 + c8 * 8],
                 Vt + vbase + (size_t)row * TT + jb * 64 + c8 * 8);
    }
  };

  // Q fragments resident in registers (pre-scaled by 1/sqrt(dk))
  v16h qf0 = frag_a(Q + kbase, q0 + r, DK, 0,  g);
  v16h qf1 = frag_a(Q + kbase, q0 + r, DK, 32, g);

  v8f o[4] = {};
  float rm[8], rs[8];
#pragma unroll
  for (int v = 0; v < 8; ++v) { rm[v] = -3.0e38f; rs[v] = 0.0f; }

  h16* Pw = Ps + wave * 16 * 64;

  issue_tile(0, 0);                               // prologue fill
  wait_async0();
  __syncthreads();
  int cur = 0;

  for (int jb = 0; jb <= qblk; ++jb) {
    if (jb < qblk) issue_tile(jb + 1, cur ^ 1);   // fetch next tile behind compute

    // S = Q * K^T for 4 key sub-tiles
    const h16* Kt = &Ks[cur][0];
    const h16* Vv = &Vs[cur][0];
    v8f s4[4];
#pragma unroll
    for (int nt = 0; nt < 4; ++nt) {
      v8f s = {};
      s = wmma32(qf0, frag_b(Kt, nt * 16 + r, 64, 0,  g), s);
      s = wmma32(qf1, frag_b(Kt, nt * 16 + r, 64, 32, g), s);
      s4[nt] = s;
    }
    // block-wide online softmax (rows m = g*8+v live across the 16-lane half-wave)
#pragma unroll
    for (int v = 0; v < 8; ++v) {
      int qq = q0 + g * 8 + v;
      float tmax = -3.0e38f;
#pragma unroll
      for (int nt = 0; nt < 4; ++nt) {
        int kk = jb * 64 + nt * 16 + r;
        float sv = (kk <= qq) ? s4[nt][v] : -1.0e9f;  // causal mask
        s4[nt][v] = sv;
        tmax = fmaxf(tmax, sv);
      }
      for (int msk = 1; msk < 16; msk <<= 1)
        tmax = fmaxf(tmax, __shfl_xor(tmax, msk, 16));
      float mnew  = fmaxf(rm[v], tmax);
      float alpha = __expf(rm[v] - mnew);
      rm[v] = mnew;
      float psum = 0.0f;
#pragma unroll
      for (int nt = 0; nt < 4; ++nt) {
        float p = __expf(s4[nt][v] - mnew);
        psum += p;
        Pw[(g * 8 + v) * 64 + nt * 16 + r] = (h16)p;
      }
      for (int msk = 1; msk < 16; msk <<= 1)
        psum += __shfl_xor(psum, msk, 16);
      rs[v] = rs[v] * alpha + psum;
#pragma unroll
      for (int dt = 0; dt < 4; ++dt) o[dt][v] *= alpha;
    }
    __syncthreads();                              // order P writes before P reads

    // O += P * V  (B operand = rows of V^T from LDS)
#pragma unroll
    for (int ks = 0; ks < 2; ++ks) {
      v16h pa = frag_a(Pw, r, 64, ks * 32, g);
#pragma unroll
      for (int dt = 0; dt < 4; ++dt) {
        v16h vb = frag_b(Vv, dt * 16 + r, 64, ks * 32, g);
        o[dt] = wmma32(pa, vb, o[dt]);
      }
    }

    wait_async0();                                // next tile's async copies landed
    __syncthreads();                              // all waves done with cur + next visible
    cur ^= 1;
  }

  // normalize and write ctx [B, T, H*dk] as f16
  size_t cbase = ((size_t)(b * TT + q0 + g * 8)) * DD + h * DK + r;
#pragma unroll
  for (int v = 0; v < 8; ++v) {
    float inv = 1.0f / rs[v];
#pragma unroll
    for (int dt = 0; dt < 4; ++dt)
      Cx[cbase + (size_t)v * DD + dt * 16] = (h16)(o[dt][v] * inv);
  }
}

// ---------------- Output projection: out = ctx @ Wo^T + bo (f32 out) ----------------
__global__ __launch_bounds__(128) void oproj_kernel(
    const h16* __restrict__ Cx, const float* __restrict__ Wo,
    const float* __restrict__ bo, float* __restrict__ out)
{
  __shared__ __align__(16) h16 as_[2][64 * 32];
  __shared__ __align__(16) h16 bs_[2][64 * 32];
  const int tid = threadIdx.x;
  const int lane = tid & 31, wave = tid >> 5;
  const int g = lane >> 4, r = lane & 15;
  const int m0 = blockIdx.x * 64;
  const int n0 = blockIdx.y * 64;

  v8h ar[2];
  float4 wr[4];
  auto load_regs = [&](int kb) {
#pragma unroll
    for (int i = 0; i < 2; ++i) {
      int idx = tid + i * 128, row = idx >> 2, c8 = idx & 3;
      ar[i] = *(const v8h*)(Cx + (size_t)(m0 + row) * DD + kb + c8 * 8);
    }
#pragma unroll
    for (int i = 0; i < 4; ++i) {
      int idx = tid + i * 128, row = idx >> 3, c4 = idx & 7;
      wr[i] = *(const float4*)(Wo + (size_t)(n0 + row) * DD + kb + c4 * 4);
    }
  };
  auto store_lds = [&](int buf) {
#pragma unroll
    for (int i = 0; i < 2; ++i) {
      int idx = tid + i * 128, row = idx >> 2, c8 = idx & 3;
      *(v8h*)(&as_[buf][row * 32 + c8 * 8]) = ar[i];
    }
#pragma unroll
    for (int i = 0; i < 4; ++i) {
      int idx = tid + i * 128, row = idx >> 3, c4 = idx & 7;
      v4h wh = { (h16)wr[i].x, (h16)wr[i].y, (h16)wr[i].z, (h16)wr[i].w };
      *(v4h*)(&bs_[buf][row * 32 + c4 * 4]) = wh;
    }
  };

  v8f acc[4] = {};
  load_regs(0);
  store_lds(0);
  __syncthreads();
  int cur = 0;
  for (int s = 0; s < 32; ++s) {
    if (s < 31) load_regs((s + 1) * 32);
    v16h a = frag_a(&as_[cur][0], wave * 16 + r, 32, 0, g);
#pragma unroll
    for (int nt = 0; nt < 4; ++nt)
      acc[nt] = wmma32(a, frag_b(&bs_[cur][0], nt * 16 + r, 32, 0, g), acc[nt]);
    if (s < 31) store_lds(cur ^ 1);
    __syncthreads();
    cur ^= 1;
  }

  float bov[4];
#pragma unroll
  for (int nt = 0; nt < 4; ++nt) bov[nt] = bo[n0 + nt * 16 + r];
  size_t base = (size_t)(m0 + wave * 16 + g * 8) * DD + n0 + r;
#pragma unroll
  for (int v = 0; v < 8; ++v)
#pragma unroll
    for (int nt = 0; nt < 4; ++nt)
      out[base + (size_t)v * DD + nt * 16] = acc[nt][v] + bov[nt];
}

extern "C" void kernel_launch(void* const* d_in, const int* in_sizes, int n_in,
                              void* d_out, int out_size, void* d_ws, size_t ws_size,
                              hipStream_t stream) {
  const float* x  = (const float*)d_in[0];
  // d_in[1] = causal mask (bool) — implemented analytically, unused
  const float* Wq = (const float*)d_in[2];
  const float* Wk = (const float*)d_in[3];
  const float* Wv = (const float*)d_in[4];
  const float* Wo = (const float*)d_in[5];
  const float* bo = (const float*)d_in[6];
  float* out = (float*)d_out;

  // workspace: Q, K, V^T, ctx — each 8192*1024 f16 = 16 MB (total 64 MB)
  h16* Q  = (h16*)d_ws;
  h16* Kc = Q  + (size_t)8192 * 1024;
  h16* Vt = Kc + (size_t)8192 * 1024;
  h16* Cx = Vt + (size_t)8192 * 1024;

  dim3 gg(128, 16), blk(128);
  qkv_kernel<0><<<gg, blk, 0, stream>>>(x, Wq, Q);
  qkv_kernel<1><<<gg, blk, 0, stream>>>(x, Wk, Kc);
  qkv_kernel<2><<<gg, blk, 0, stream>>>(x, Wv, Vt);
  attn_kernel <<<dim3(32, 64), blk, 0, stream>>>(Q, Kc, Vt, Cx);
  oproj_kernel<<<gg, blk, 0, stream>>>(Cx, Wo, bo, out);
}
